// MLPRoPE_54760833024720
// MI455X (gfx1250) — matrix operand order repaired and checked by reading, not compiled, not run on hardware
//
#include <hip/hip_runtime.h>
#include <hip/hip_bf16.h>

typedef __attribute__((ext_vector_type(16))) __bf16 v16bf;
typedef __attribute__((ext_vector_type(8)))  float  v8f;
typedef unsigned short u16;

// ---------------------------------------------------------------------------
// Problem constants (from reference setup_inputs)
// ---------------------------------------------------------------------------
#define TT 4
#define BB 32
#define LL 196
#define DD 384
#define HH 1536
#define MM (BB * TT * LL)   // 25088 GEMM rows

// ---------------------------------------------------------------------------
// helpers
// ---------------------------------------------------------------------------
__device__ __forceinline__ u16 f2bf(float f) {
    unsigned u = __float_as_uint(f);
    unsigned r = u + 0x7FFFu + ((u >> 16) & 1u);   // round-to-nearest-even
    return (u16)(r >> 16);
}

union Frag { v16bf v; uint4 q[2]; unsigned u[8]; };

// ---------------------------------------------------------------------------
// Pack x [T,B,L,D] f32  ->  xb [B, T*L, D] bf16   (transpose(0,1) + flatten)
// ---------------------------------------------------------------------------
__global__ void snn_pack_x(const float* __restrict__ x, u16* __restrict__ xb,
                           long total) {
    long idx = (long)blockIdx.x * blockDim.x + threadIdx.x;
    if (idx >= total) return;
    int d = (int)(idx % DD);
    long r = idx / DD;
    int l = (int)(r % LL); r /= LL;
    int b = (int)(r % BB);
    int t = (int)(r / BB);
    long out = (((long)b * TT + t) * LL + l) * DD + d;
    xb[out] = f2bf(x[idx]);
}

// ---------------------------------------------------------------------------
// Transpose weight [K,N] f32 -> [N,K] bf16  (so GEMM B K-pairs are contiguous)
// ---------------------------------------------------------------------------
__global__ void snn_pack_wt(const float* __restrict__ W, u16* __restrict__ Wt,
                            int K, int N) {
    long idx = (long)blockIdx.x * blockDim.x + threadIdx.x;
    if (idx >= (long)K * N) return;
    int n = (int)(idx % N);
    int k = (int)(idx / N);
    Wt[(long)n * K + k] = f2bf(W[idx]);
}

// ---------------------------------------------------------------------------
// WMMA GEMM, LDS-staged with CDNA5 async global->LDS copies (ASYNCcnt):
//   C[M,N] = A[M,K](bf16) * Bt[N,K](bf16)^T + bias
// Block = 256 threads (8 waves), block tile 128x128, K-step 32, double-buffered
// LDS. Waves arranged 2(M) x 4(N); each wave computes a 64x32 tile = 8 WMMAs
// per K-step. A/B fragments use contiguous k = 16*(lane/16) + s; C/D per CDNA5 ISA 7.12.2.
// ---------------------------------------------------------------------------
__global__ void __launch_bounds__(256)
snn_gemm_bf16(const u16* __restrict__ A, const u16* __restrict__ Bt,
              const float* __restrict__ bias, float* __restrict__ C,
              int M, int N, int K) {
    __shared__ __align__(16) u16 shA[2][128 * 32];
    __shared__ __align__(16) u16 shB[2][128 * 32];

    const int lane = threadIdx.x;            // 0..31
    const int wave = threadIdx.y;            // 0..7
    const int tid  = wave * 32 + lane;       // 0..255

    const int tilesN = N >> 7;
    const int tm = (int)(blockIdx.x / tilesN) << 7;
    const int tn = (int)(blockIdx.x % tilesN) << 7;

    // --- staging: each thread owns two 16B chunks of A-tile and of B-tile ---
    // chunk id c in [0,512): row = c/4 (128 rows), col8 = (c%4)*8 (32 elems/row)
    const int c0 = tid * 2, c1 = tid * 2 + 1;
    const int r0 = c0 >> 2, e0 = (c0 & 3) * 8;
    const int r1 = c1 >> 2, e1 = (c1 & 3) * 8;
    const u16* gA0 = A  + (size_t)(tm + r0) * K + e0;
    const u16* gA1 = A  + (size_t)(tm + r1) * K + e1;
    const u16* gB0 = Bt + (size_t)(tn + r0) * K + e0;
    const u16* gB1 = Bt + (size_t)(tn + r1) * K + e1;
    const unsigned lA0 = (unsigned)(size_t)&shA[0][r0 * 32 + e0];
    const unsigned lA1 = (unsigned)(size_t)&shA[0][r1 * 32 + e1];
    const unsigned lB0 = (unsigned)(size_t)&shB[0][r0 * 32 + e0];
    const unsigned lB1 = (unsigned)(size_t)&shB[0][r1 * 32 + e1];
    const unsigned bufB = 128 * 32 * 2;      // bytes per LDS buffer

    auto issue = [&](int ks, int buf) {
        const size_t ko = (size_t)ks * 32;
        const unsigned bo = (unsigned)buf * bufB;
        asm volatile("global_load_async_to_lds_b128 %0, %1, off"
                     :: "v"(lA0 + bo),
                        "v"((unsigned long long)(size_t)(gA0 + ko)) : "memory");
        asm volatile("global_load_async_to_lds_b128 %0, %1, off"
                     :: "v"(lA1 + bo),
                        "v"((unsigned long long)(size_t)(gA1 + ko)) : "memory");
        asm volatile("global_load_async_to_lds_b128 %0, %1, off"
                     :: "v"(lB0 + bo),
                        "v"((unsigned long long)(size_t)(gB0 + ko)) : "memory");
        asm volatile("global_load_async_to_lds_b128 %0, %1, off"
                     :: "v"(lB1 + bo),
                        "v"((unsigned long long)(size_t)(gB1 + ko)) : "memory");
    };

    // --- compute mapping ---
    const int g  = lane >> 4;                // K-half select
    const int h  = lane & 15;                // row/col within 16
    const int wm = wave >> 2;                // 0..1  (64-row slab)
    const int wn = wave & 3;                 // 0..3  (32-col slab)

    v8f acc[4][2] = {};
    const int nk = K >> 5;

    issue(0, 0);
    for (int ks = 0; ks < nk; ++ks) {
        const int buf = ks & 1;
        asm volatile("s_wait_asynccnt 0x0" ::: "memory");
        __syncthreads();
        if (ks + 1 < nk) issue(ks + 1, buf ^ 1);

        const u16* sA = shA[buf];
        const u16* sB = shB[buf];

        Frag bf[2];
#pragma unroll
        for (int ni = 0; ni < 2; ++ni) {
            // B 32x16 layout: VGPR i holds K=16g+2i..+1 for col = lane%16
            const u16* p = sB + ((wn * 32 + ni * 16 + h) * 32 + 16 * g);
            bf[ni].q[0] = *(const uint4*)p;          // K elems 16g..16g+7
            bf[ni].q[1] = *(const uint4*)(p + 8);    // K elems 16g+8..16g+15
        }
#pragma unroll
        for (int mi = 0; mi < 4; ++mi) {
            // A 16x32 layout: VGPR i holds K=16g+2i..+1 for row = lane%16
            Frag af;
            const u16* p = sA + ((wm * 64 + mi * 16 + h) * 32 + 16 * g);
            af.q[0] = *(const uint4*)p;              // K elems 16g..16g+7
            af.q[1] = *(const uint4*)(p + 8);        // K elems 16g+8..16g+15
            acc[mi][0] = __builtin_amdgcn_wmma_f32_16x16x32_bf16(
                false, af.v, false, bf[0].v, (short)0, acc[mi][0], false, false);
            acc[mi][1] = __builtin_amdgcn_wmma_f32_16x16x32_bf16(
                false, af.v, false, bf[1].v, (short)0, acc[mi][1], false, false);
        }
    }

    // C/D f32 layout: VGPR r -> row r + 8*g, col = lane%16
#pragma unroll
    for (int mi = 0; mi < 4; ++mi) {
        const int rbase = tm + wm * 64 + mi * 16 + g * 8;
#pragma unroll
        for (int ni = 0; ni < 2; ++ni) {
            const int cc = tn + wn * 32 + ni * 16 + h;
            const float bi = bias[cc];
#pragma unroll
            for (int r = 0; r < 8; ++r)
                C[(size_t)(rbase + r) * N + cc] = acc[mi][ni][r] + bi;
        }
    }
}

// ---------------------------------------------------------------------------
// BatchNorm statistics: per channel over all rows. One thread per channel
// (coalesced across the block), f32 accumulation.
// ---------------------------------------------------------------------------
__global__ void snn_bn_stats(const float* __restrict__ X, int rows, int C,
                             float* __restrict__ mean, float* __restrict__ rstd) {
    int c = blockIdx.x * blockDim.x + threadIdx.x;
    if (c >= C) return;
    float s = 0.f, ss = 0.f;
    for (int r = 0; r < rows; ++r) {
        float v = X[(long)r * C + c];
        s += v; ss += v * v;
    }
    float m = s / rows;
    mean[c] = m;
    rstd[c] = rsqrtf(ss / rows - m * m + 1e-5f);
}

// ---------------------------------------------------------------------------
// BN apply + LIF (tau=2, v_th=1, hard reset) -> spikes as bf16 {0,1}
// H layout [B, T*L, C]; one thread per (b,l,c), sequential over T.
// ---------------------------------------------------------------------------
__global__ void snn_bn_lif_spike(const float* __restrict__ H,
                                 const float* __restrict__ mean,
                                 const float* __restrict__ rstd,
                                 const float* __restrict__ gamma,
                                 const float* __restrict__ beta,
                                 u16* __restrict__ S, int C, long total) {
    long idx = (long)blockIdx.x * blockDim.x + threadIdx.x;
    if (idx >= total) return;
    int c = (int)(idx % C);
    long bl = idx / C;
    int l = (int)(bl % LL);
    int b = (int)(bl / LL);
    const float m = mean[c], rs = rstd[c], ga = gamma[c], be = beta[c];
    long base = (((long)b * TT) * LL + l) * C + c;     // t stride = L*C
    const long ts = (long)LL * C;
    float v = 0.f;
#pragma unroll
    for (int t = 0; t < TT; ++t) {
        float x = (H[base + t * ts] - m) * rs * ga + be;
        v = 0.5f * (v + x);                 // v += (x - v)/tau, tau = 2
        float sp = (v >= 1.0f) ? 1.0f : 0.0f;
        v *= (1.0f - sp);                   // hard reset to 0
        S[base + t * ts] = (sp != 0.f) ? (u16)0x3F80 : (u16)0;
    }
}

// ---------------------------------------------------------------------------
// BN apply + LIF -> final f32 output in [T, B, L, D] order
// ---------------------------------------------------------------------------
__global__ void snn_bn_lif_out(const float* __restrict__ H,
                               const float* __restrict__ mean,
                               const float* __restrict__ rstd,
                               const float* __restrict__ gamma,
                               const float* __restrict__ beta,
                               float* __restrict__ O, int C, long total) {
    long idx = (long)blockIdx.x * blockDim.x + threadIdx.x;
    if (idx >= total) return;
    int c = (int)(idx % C);
    long bl = idx / C;
    int l = (int)(bl % LL);
    int b = (int)(bl / LL);
    const float m = mean[c], rs = rstd[c], ga = gamma[c], be = beta[c];
    long base = (((long)b * TT) * LL + l) * C + c;
    const long ts = (long)LL * C;
    float v = 0.f;
#pragma unroll
    for (int t = 0; t < TT; ++t) {
        float x = (H[base + t * ts] - m) * rs * ga + be;
        v = 0.5f * (v + x);
        float sp = (v >= 1.0f) ? 1.0f : 0.0f;
        v *= (1.0f - sp);
        O[(((long)t * BB + b) * LL + l) * C + c] = sp;
    }
}

// ---------------------------------------------------------------------------
// launch
// ---------------------------------------------------------------------------
extern "C" void kernel_launch(void* const* d_in, const int* in_sizes, int n_in,
                              void* d_out, int out_size, void* d_ws, size_t ws_size,
                              hipStream_t stream) {
    const float* x      = (const float*)d_in[0];
    const float* W1     = (const float*)d_in[1];
    const float* b1     = (const float*)d_in[2];
    const float* gamma1 = (const float*)d_in[3];
    const float* beta1  = (const float*)d_in[4];
    const float* W2     = (const float*)d_in[5];
    const float* b2     = (const float*)d_in[6];
    const float* gamma2 = (const float*)d_in[7];
    const float* beta2  = (const float*)d_in[8];
    float* out = (float*)d_out;

    // workspace carve (256B aligned)
    size_t off = 0;
    auto carve = [&](size_t bytes) -> void* {
        void* p = (char*)d_ws + off;
        off += (bytes + 255) & ~(size_t)255;
        return p;
    };
    u16*   xb    = (u16*)  carve((size_t)MM * DD * 2);   // bf16 A for GEMM1
    u16*   w1t   = (u16*)  carve((size_t)HH * DD * 2);   // W1^T bf16 [H,D]
    u16*   w2t   = (u16*)  carve((size_t)DD * HH * 2);   // W2^T bf16 [D,H]
    float* h1    = (float*)carve((size_t)MM * HH * 4);   // GEMM1 out
    u16*   spk   = (u16*)  carve((size_t)MM * HH * 2);   // bf16 spikes (A for GEMM2)
    float* h2    = (float*)carve((size_t)MM * DD * 4);   // GEMM2 out
    float* mean1 = (float*)carve(HH * 4);
    float* rstd1 = (float*)carve(HH * 4);
    float* mean2 = (float*)carve(DD * 4);
    float* rstd2 = (float*)carve(DD * 4);
    (void)ws_size; (void)in_sizes; (void)n_in; (void)out_size;

    const long nX = (long)MM * DD;          // 9,633,792
    const long nH = (long)BB * LL * HH;
    const long nD = (long)BB * LL * DD;

    // 1) pack inputs to bf16
    snn_pack_x <<<dim3((unsigned)((nX + 255) / 256)), dim3(256), 0, stream>>>(x, xb, nX);
    snn_pack_wt<<<dim3((unsigned)(((long)DD * HH + 255) / 256)), dim3(256), 0, stream>>>(W1, w1t, DD, HH);
    snn_pack_wt<<<dim3((unsigned)(((long)HH * DD + 255) / 256)), dim3(256), 0, stream>>>(W2, w2t, HH, DD);

    // 2) GEMM1: [M,384] x [384,1536] -> h1   (128x128 tiles: 196 x 12 blocks)
    snn_gemm_bf16<<<dim3((MM / 128) * (HH / 128)), dim3(32, 8), 0, stream>>>(
        xb, w1t, b1, h1, MM, HH, DD);

    // 3) BN1 stats + BN+LIF -> spikes
    snn_bn_stats<<<dim3((HH + 255) / 256), dim3(256), 0, stream>>>(h1, MM, HH, mean1, rstd1);
    snn_bn_lif_spike<<<dim3((unsigned)((nH + 255) / 256)), dim3(256), 0, stream>>>(
        h1, mean1, rstd1, gamma1, beta1, spk, HH, nH);

    // 4) GEMM2: [M,1536] x [1536,384] -> h2  (196 x 3 blocks)
    snn_gemm_bf16<<<dim3((MM / 128) * (DD / 128)), dim3(32, 8), 0, stream>>>(
        spk, w2t, b2, h2, MM, DD, HH);

    // 5) BN2 stats + BN+LIF -> output [T,B,L,D] f32
    snn_bn_stats<<<dim3((DD + 255) / 256), dim3(256), 0, stream>>>(h2, MM, DD, mean2, rstd2);
    snn_bn_lif_out<<<dim3((unsigned)((nD + 255) / 256)), dim3(256), 0, stream>>>(
        h2, mean2, rstd2, gamma2, beta2, out, DD, nD);
}
